// Complex_LSTM_42159398977724
// MI455X (gfx1250) — matrix operand order repaired and checked by reading, not compiled, not run on hardware
//
#include <hip/hip_runtime.h>
#include <stdint.h>

// ---------------------------------------------------------------------------
// Complex LSTM for MI455X (gfx1250), wave32 + WMMA bf16.
//   Phase 0: pack W,U (f32 -> bf16) into per-lane WMMA B-tile layout.
//   Phase 1: zx = x@W + b  (complex GEMM, bf16 WMMA, f32 accum) -> ws.
//   Phase 2: persistent recurrent kernel, 32 blocks x 256 thr (8 waves =
//            4 gates x 2 K-halves), U-slice LDS-resident (128KB, staged with
//            global_load_async_to_lds_b128), grid-wide barrier per step.
// Workspace requirement: ~533 MB.
// ---------------------------------------------------------------------------

typedef __attribute__((ext_vector_type(16))) __bf16 v16bf;
typedef __attribute__((ext_vector_type(8)))  __bf16 v8bf;
typedef __attribute__((ext_vector_type(8)))  float  v8f;

#define B_   16
#define T_   2000
#define N_   512
#define U_   512
#define FU_  2048
#define NWG_ 32                       // U_/16 column-tile workgroups in phase 2
static const size_t PKCOMP = (size_t)(FU_ / 16) * 16 * 32 * 16; // 1,048,576 elems / component

__device__ __forceinline__ v8f wmma_bf(v16bf a, v16bf b, v8f c) {
  return __builtin_amdgcn_wmma_f32_16x16x32_bf16(false, a, false, b, (short)0, c, false, false);
}
__device__ __forceinline__ v16bf cat16(v8bf lo, v8bf hi) {
  return __builtin_shufflevector(lo, hi, 0,1,2,3,4,5,6,7,8,9,10,11,12,13,14,15);
}
__device__ __forceinline__ v8bf ld8bf(const __bf16* p) { return *(const v8bf*)p; }
__device__ __forceinline__ v8bf cvt8(const float* p) {
  float4 a = *(const float4*)p, b = *(const float4*)(p + 4);
  v8bf r;
  r[0]=(__bf16)a.x; r[1]=(__bf16)a.y; r[2]=(__bf16)a.z; r[3]=(__bf16)a.w;
  r[4]=(__bf16)b.x; r[5]=(__bf16)b.y; r[6]=(__bf16)b.z; r[7]=(__bf16)b.w;
  return r;
}
__device__ __forceinline__ float sigm(float x) { return 1.0f / (1.0f + __expf(-x)); }
__device__ __forceinline__ v8f vzero8() {
  v8f z = {0.f,0.f,0.f,0.f,0.f,0.f,0.f,0.f};
  return z;
}

// ---------------------------------------------------------------------------
// Phase 0: pack f32 [K=512][N=2048] matrices into bf16 WMMA-B tiles:
//   pk[comp][nt][kt][lane][e],  K = kt*32 + (lane>>4)*16 + e,  N = nt*16 + (lane&15)
// ---------------------------------------------------------------------------
__global__ void pack_weights(const float* __restrict__ Wr, const float* __restrict__ Wi,
                             const float* __restrict__ Ur, const float* __restrict__ Ui,
                             __bf16* __restrict__ Wpk, __bf16* __restrict__ Upk) {
  size_t id = (size_t)blockIdx.x * blockDim.x + threadIdx.x;   // 4 * PKCOMP threads
  int a = (int)(id >> 20);
  size_t rem = id & (PKCOMP - 1);
  int e    = (int)(rem & 15);
  int lane = (int)((rem >> 4) & 31);
  int kt   = (int)((rem >> 9) & 15);
  int nt   = (int)(rem >> 13);
  int K = kt * 32 + (lane >> 4) * 16 + e;
  int Nc = nt * 16 + (lane & 15);
  const float* src = (a == 0) ? Wr : (a == 1) ? Wi : (a == 2) ? Ur : Ui;
  __bf16* dst = ((a < 2) ? Wpk : Upk) + (size_t)(a & 1) * PKCOMP;
  dst[rem] = (__bf16)src[(size_t)K * FU_ + Nc];
}

// ---------------------------------------------------------------------------
// Phase 1: zx = x@W + b (complex). One wave computes a 16(M) x 64(N) block.
// zx layout: [t][col][b][{re,im}] f32, so phase-2 reads are contiguous per lane.
// ---------------------------------------------------------------------------
__global__ __launch_bounds__(32)
void gemm_in(const float* __restrict__ xr, const float* __restrict__ xi,
             const __bf16* __restrict__ Wpk,
             const float* __restrict__ br, const float* __restrict__ bi,
             float2* __restrict__ zx) {
  const int l = threadIdx.x, half = l >> 4, lm = l & 15;
  const int mt = blockIdx.x;          // 0..1999 row tiles (2000 = 125 tiles * 16 batches)
  const int nb = blockIdx.y;          // 0..31, each = 4 column tiles
  const int b  = mt / 125;
  const int tb = (mt % 125) * 16;

  const float* xrow_r = xr + (size_t)(b * T_ + tb + lm) * N_;
  const float* xrow_i = xi + (size_t)(b * T_ + tb + lm) * N_;

  v8f acc[4][4];                       // [ntile][rr,ii,ri,ir]
#pragma unroll
  for (int j = 0; j < 4; ++j)
#pragma unroll
    for (int k = 0; k < 4; ++k) acc[j][k] = vzero8();

  for (int kt = 0; kt < 16; ++kt) {
    const int k0 = kt * 32;
    v16bf Ar = cat16(cvt8(xrow_r + k0 + half * 16), cvt8(xrow_r + k0 + half * 16 + 8));
    v16bf Ai = cat16(cvt8(xrow_i + k0 + half * 16), cvt8(xrow_i + k0 + half * 16 + 8));
#pragma unroll
    for (int j = 0; j < 4; ++j) {
      const int nt = nb * 4 + j;
      const size_t boff = ((size_t)(nt * 16 + kt) * 32 + l) * 16;
      v16bf Br = *(const v16bf*)(Wpk + boff);
      v16bf Bi = *(const v16bf*)(Wpk + PKCOMP + boff);
      acc[j][0] = wmma_bf(Ar, Br, acc[j][0]);
      acc[j][1] = wmma_bf(Ai, Bi, acc[j][1]);
      acc[j][2] = wmma_bf(Ar, Bi, acc[j][2]);
      acc[j][3] = wmma_bf(Ai, Br, acc[j][3]);
    }
  }

#pragma unroll
  for (int j = 0; j < 4; ++j) {
    const int col = (nb * 4 + j) * 16 + lm;
    const float brv = br[col], biv = bi[col];
#pragma unroll
    for (int v = 0; v < 8; ++v) {
      const int t = tb + v + half * 8;                 // C layout: m = v + half*8
      const float zrv = acc[j][0][v] - acc[j][1][v] + brv;
      const float ziv = acc[j][2][v] + acc[j][3][v] + biv;
      zx[((size_t)t * FU_ + col) * 16 + b] = make_float2(zrv, ziv);
    }
  }
}

// ---------------------------------------------------------------------------
// Phase 2: persistent recurrent kernel. 32 blocks x 256 threads (8 waves).
// Wave w: gate g = w&3, K-half kh = w>>2 (kt in [kh*8, kh*8+8)).
// Block owns h columns [ut*16, ut*16+16).
// ---------------------------------------------------------------------------
__global__ __launch_bounds__(256)
void lstm_rec(const __bf16* __restrict__ Upk, const float* __restrict__ zx,
              __bf16* __restrict__ hbuf, int* __restrict__ syncc,
              float* __restrict__ out) {
  __shared__ __align__(32) __bf16 sB[4 * 2 * 8192];   // 128 KB: U slice, [gate][comp] tiles
  __shared__ float sZ[2 * 8 * 256];                   // [kh][gate*2+comp][cell] partials, 16 KB
  __shared__ float sC[2 * 256];                       // cell state (re/im)

  const int tid = threadIdx.x, l = tid & 31, w = tid >> 5;
  const int g = w & 3, kh = w >> 2;
  const int half = l >> 4, lm = l & 15;
  const int ut = blockIdx.x, u0 = ut * 16;

  // One-time: stage this block's U slice into LDS via the CDNA5 async path
  // (no VGPR round-trip). Low 32 bits of a flat __shared__ pointer == LDS offset.
  {
    const uint32_t lds0 = (uint32_t)(size_t)(void*)&sB[0];
    for (int i = tid; i < 8192; i += 256) {            // 8192 x 16B chunks = 128 KB
      const int gc = i >> 10, off = i & 1023;          // gc = gate*2+comp
      const __bf16* src = Upk + (size_t)(gc & 1) * PKCOMP +
                          (size_t)((gc >> 1) * 32 + ut) * 8192 + (size_t)off * 8;
      uint64_t ga = (uint64_t)(size_t)src;
      uint32_t la = lds0 + (uint32_t)i * 16u;
      asm volatile("global_load_async_to_lds_b128 %0, %1, off"
                   :: "v"(la), "v"(ga) : "memory");
    }
    asm volatile("s_wait_asynccnt 0x0" ::: "memory");
  }
  for (int i = tid; i < 512; i += 256) sC[i] = 0.f;
  __syncthreads();

  const __bf16* sBr = sB + (g * 2 + 0) * 8192;
  const __bf16* sBi = sB + (g * 2 + 1) * 8192;
  const int col = g * 512 + u0 + lm;

  for (int t = 0; t < T_; ++t) {
    const int buf = t & 1;
    const __bf16* hr = hbuf + (size_t)(buf * 2 + 0) * (B_ * U_) + lm * U_;  // A row = batch = lm
    const __bf16* hi = hbuf + (size_t)(buf * 2 + 1) * (B_ * U_) + lm * U_;

    if (t + 1 < T_)  // pull next step's zx tile toward the caches (global_prefetch_b8)
      __builtin_prefetch(zx + (((size_t)(t + 1) * FU_ + col) * 16 + half * 8) * 2, 0, 1);

    v8f rr = vzero8(), ii2 = vzero8(), ri = vzero8(), ir = vzero8();
#pragma unroll 4
    for (int kt = kh * 8; kt < kh * 8 + 8; ++kt) {
      const int k0 = kt * 32;
      v16bf Ar = cat16(ld8bf(hr + k0 + half * 16), ld8bf(hr + k0 + half * 16 + 8));
      v16bf Ai = cat16(ld8bf(hi + k0 + half * 16), ld8bf(hi + k0 + half * 16 + 8));
      v16bf Br = *(const v16bf*)(sBr + (kt * 32 + l) * 16);
      v16bf Bi = *(const v16bf*)(sBi + (kt * 32 + l) * 16);
      rr  = wmma_bf(Ar, Br, rr);
      ii2 = wmma_bf(Ai, Bi, ii2);
      ri  = wmma_bf(Ar, Bi, ri);
      ir  = wmma_bf(Ai, Br, ir);
    }

    // Partial z (this K-half); kh==0 wave folds in zx_t + bias (already in zx).
    float zxr[8] = {0,0,0,0,0,0,0,0}, zxi[8] = {0,0,0,0,0,0,0,0};
    if (kh == 0) {
      const float4* zp = (const float4*)(zx + (((size_t)t * FU_ + col) * 16 + half * 8) * 2);
      float4 z0 = zp[0], z1 = zp[1], z2 = zp[2], z3 = zp[3];
      zxr[0]=z0.x; zxr[1]=z0.z; zxr[2]=z1.x; zxr[3]=z1.z; zxr[4]=z2.x; zxr[5]=z2.z; zxr[6]=z3.x; zxr[7]=z3.z;
      zxi[0]=z0.y; zxi[1]=z0.w; zxi[2]=z1.y; zxi[3]=z1.w; zxi[4]=z2.y; zxi[5]=z2.w; zxi[6]=z3.y; zxi[7]=z3.w;
    }
#pragma unroll
    for (int v = 0; v < 8; ++v) {
      const int m = v + half * 8;
      const int cell = m * 16 + lm;
      sZ[(kh * 8 + g * 2 + 0) * 256 + cell] = rr[v] - ii2[v] + zxr[v];
      sZ[(kh * 8 + g * 2 + 1) * 256 + cell] = ri[v] + ir[v] + zxi[v];
    }
    __syncthreads();

    // Gate math: 256 (batch, col) cells, 1 per thread; split activations on re/im.
    const int nb = (t + 1) & 1;
    {
      const int cell = tid;
      const int m = cell >> 4, u = cell & 15;
      float zv[8];
#pragma unroll
      for (int q = 0; q < 8; ++q)
        zv[q] = sZ[q * 256 + cell] + sZ[(8 + q) * 256 + cell];   // combine K-halves
      const float ar = tanhf(zv[0]), ai = tanhf(zv[1]);
      const float gir = sigm(zv[2]), gii = sigm(zv[3]);
      const float fr = sigm(zv[4]),  fi = sigm(zv[5]);
      const float og = sigm(zv[6]),  oi = sigm(zv[7]);
      const float cr = sC[cell], ci = sC[256 + cell];
      const float ncr = (ar * gir - ai * gii) + (fr * cr - fi * ci);
      const float nci = (ar * gii + ai * gir) + (fr * ci + fi * cr);
      sC[cell] = ncr; sC[256 + cell] = nci;
      const float tr = tanhf(ncr), ti = tanhf(nci);
      const float hrv = og * tr - oi * ti;
      const float hiv = og * ti + oi * tr;
      const size_t o = (((size_t)m * T_ + t) * U_ + (u0 + u)) * 2;
      out[o] = hrv; out[o + 1] = hiv;
      hbuf[(size_t)(nb * 2 + 0) * (B_ * U_) + m * U_ + u0 + u] = (__bf16)hrv;
      hbuf[(size_t)(nb * 2 + 1) * (B_ * U_) + m * U_ + u0 + u] = (__bf16)hiv;
    }

    // Device-wide barrier. h is double-buffered so one phase suffices.
    __syncthreads();            // all threads done computing/storing this step
    __threadfence();            // release: every wave flushes its own stores
    __syncthreads();            // all release fences complete before arrive
    if (tid == 0) {
      atomicAdd(syncc, 1);
      const int target = NWG_ * (t + 1);
      while (__hip_atomic_load(syncc, __ATOMIC_RELAXED, __HIP_MEMORY_SCOPE_AGENT) < target)
        __builtin_amdgcn_s_sleep(1);
    }
    __syncthreads();            // everyone past the barrier
    __threadfence();            // acquire: every wave invalidates before reading new h
  }
}

// ---------------------------------------------------------------------------
extern "C" void kernel_launch(void* const* d_in, const int* in_sizes, int n_in,
                              void* d_out, int out_size, void* d_ws, size_t ws_size,
                              hipStream_t stream) {
  const float* xr = (const float*)d_in[0];
  const float* xi = (const float*)d_in[1];
  const float* Wr = (const float*)d_in[2];
  const float* Wi = (const float*)d_in[3];
  const float* Ur = (const float*)d_in[4];
  const float* Ui = (const float*)d_in[5];
  const float* br = (const float*)d_in[6];
  const float* bi = (const float*)d_in[7];

  uint8_t* w = (uint8_t*)d_ws;
  auto alignup = [](size_t x) { return (x + 255) & ~(size_t)255; };
  size_t off = 0;
  __bf16* Wpk = (__bf16*)(w + off); off = alignup(off + 2 * PKCOMP * sizeof(__bf16));   // 4 MB
  __bf16* Upk = (__bf16*)(w + off); off = alignup(off + 2 * PKCOMP * sizeof(__bf16));   // 4 MB
  float*  zx  = (float*)(w + off);  off = alignup(off + (size_t)T_ * FU_ * B_ * 2 * sizeof(float)); // 524 MB
  __bf16* hbuf = (__bf16*)(w + off); off = alignup(off + (size_t)2 * 2 * B_ * U_ * sizeof(__bf16)); // 64 KB
  int* syncc = (int*)(w + off); off += 256;
  (void)ws_size; (void)in_sizes; (void)n_in; (void)out_size;

  hipMemsetAsync(hbuf, 0, (size_t)2 * 2 * B_ * U_ * sizeof(__bf16), stream); // h0 = 0 (both buffers)
  hipMemsetAsync(syncc, 0, 256, stream);                                     // barrier counter = 0

  pack_weights<<<(unsigned)((4 * PKCOMP) / 256), 256, 0, stream>>>(Wr, Wi, Ur, Ui, Wpk, Upk);
  gemm_in<<<dim3(2000, 32), 32, 0, stream>>>(xr, xi, Wpk, br, bi, (float2*)zx);
  lstm_rec<<<NWG_, 256, 0, stream>>>(Upk, zx, hbuf, syncc, (float*)d_out);
}
